// MSDeformableAttention3D_13932873909053
// MI455X (gfx1250) — matrix operand order repaired and checked by reading, not compiled, not run on hardware
//
#include <hip/hip_runtime.h>

#define HEADS  8
#define LEVELS 4
#define POINTS 8
#define HD     32
#define EMBED  256
#define KDIM   256
#define LDST   20   // padded LDS row stride in uints (80B): 16B-aligned, bank-spread

typedef __attribute__((ext_vector_type(16))) __bf16        v16bf;
typedef __attribute__((ext_vector_type(8)))  float         v8f;
typedef __attribute__((ext_vector_type(8)))  unsigned int  v8u;

// ---- fp32 -> bf16 (round-to-nearest-even) ----------------------------------
__device__ __forceinline__ unsigned short f2bf(float v) {
  unsigned int a = __float_as_uint(v);
  a = (a + 0x7FFFu + ((a >> 16) & 1u)) >> 16;
  return (unsigned short)a;
}
__device__ __forceinline__ unsigned int pack2bf(float lo, float hi) {
  return (unsigned int)f2bf(lo) | ((unsigned int)f2bf(hi) << 16);
}

// Build a v16bf WMMA operand from two 16B LDS chunks (4 + 4 packed pairs).
__device__ __forceinline__ v16bf frag_from(const unsigned int* p0,
                                           const unsigned int* p1) {
  uint4 lo = *(const uint4*)p0;
  uint4 hi = *(const uint4*)p1;
  v8u u;
  u[0] = lo.x; u[1] = lo.y; u[2] = lo.z; u[3] = lo.w;
  u[4] = hi.x; u[5] = hi.y; u[6] = hi.z; u[7] = hi.w;
  return __builtin_bit_cast(v16bf, u);
}

// ---- precision/layout conversion kernels (run once per launch) -------------
// Activations: fp32 [M x 256] -> bf16 [Mpad x 256], zero-padded rows.
__global__ __launch_bounds__(256)
void cvt_act_bf16(const float* __restrict__ in, unsigned int* __restrict__ out,
                  int M, long long npairs)
{
  long long i = (long long)blockIdx.x * 256 + threadIdx.x;  // pair index
  if (i >= npairs) return;
  int m  = (int)(i >> 7);
  int kp = (int)(i & 127);
  float x0 = 0.f, x1 = 0.f;
  if (m < M) {
    const float2 f = *(const float2*)(in + (size_t)m * 256 + kp * 2);
    x0 = f.x; x1 = f.y;
  }
  out[i] = pack2bf(x0, x1);
}

// Weights: fp32 [256 x N] -> bf16 transposed [N x 256] (K contiguous per row).
__global__ __launch_bounds__(256)
void cvt_w_bf16(const float* __restrict__ W, unsigned int* __restrict__ Wt, int N)
{
  int i = blockIdx.x * 256 + threadIdx.x;   // over N*128 pairs
  if (i >= N * 128) return;
  int n = i >> 7, kp = i & 127;
  float lo = W[(size_t)(2 * kp)     * N + n];
  float hi = W[(size_t)(2 * kp + 1) * N + n];
  Wt[(size_t)n * 128 + kp] = pack2bf(lo, hi);
}

__global__ __launch_bounds__(256)
void zero_pairs(unsigned int* __restrict__ p, int count)
{
  int i = blockIdx.x * 256 + threadIdx.x;
  if (i < count) p[i] = 0;
}

// ---- async double-buffered staging ----------------------------------------
// Copies one K-step tile: A 64x32 bf16 (row-major) and B(=W^T) 128x32 bf16
// into padded LDS buffers via GLOBAL_LOAD_ASYNC_TO_LDS_B128 (ASYNCcnt).
// Per thread: 2 A-chunks + 4 B-chunks = 6 async ops.
__device__ __forceinline__ void async_copy_tile(
    const char* __restrict__ Ab, const char* __restrict__ Bb,
    unsigned ldsA, unsigned ldsB, int t, int rowBase, int colBase, int k0)
{
  #pragma unroll
  for (int i = 0; i < 2; ++i) {
    int c = t + i * 128;                     // 0..255 : 64 rows x 4 chunks
    int m = c >> 2, cc = (c & 3) * 16;
    unsigned long long ga = (unsigned long long)(uintptr_t)
        (Ab + ((size_t)(rowBase + m) * 256 + k0) * 2 + cc);
    unsigned la = ldsA + (unsigned)(m * (LDST * 4) + cc);
    asm volatile("global_load_async_to_lds_b128 %0, %1, off"
                 :: "v"(la), "v"(ga) : "memory");
  }
  #pragma unroll
  for (int i = 0; i < 4; ++i) {
    int c = t + i * 128;                     // 0..511 : 128 cols x 4 chunks
    int n = c >> 2, cc = (c & 3) * 16;
    unsigned long long gb = (unsigned long long)(uintptr_t)
        (Bb + ((size_t)(colBase + n) * 256 + k0) * 2 + cc);
    unsigned lb = ldsB + (unsigned)(n * (LDST * 4) + cc);
    asm volatile("global_load_async_to_lds_b128 %0, %1, off"
                 :: "v"(lb), "v"(gb) : "memory");
  }
}

// ----------------------------------------------------------------------------
// C[M x N] = Abf[Mpad x 256](bf16) * Wt[N x 256](bf16, K-major) (+bias)(+R)
// Block: 128 thr = 4 waves (2x2); block tile 64x128; wave tile 32x64 = 2x4 WMMA.
// Grid: (N/128, Mpad/64). Double-buffered LDS fed by async-to-LDS copies.
// ----------------------------------------------------------------------------
__global__ __launch_bounds__(128)
void gemm_bf16wmma_async(const unsigned short* __restrict__ Abf,
                         const unsigned short* __restrict__ Wt,
                         const float* __restrict__ bias,
                         const float* __restrict__ R,
                         float* __restrict__ C, int M, int ldC)
{
  __shared__ __align__(16) unsigned int Asp[2][64  * LDST];
  __shared__ __align__(16) unsigned int Bsp[2][128 * LDST];

  const int t    = threadIdx.x;
  const int lane = t & 31;
  const int wv   = t >> 5;
  const int wm   = wv >> 1, wn = wv & 1;
  const int half = lane >> 4, l15 = lane & 15;
  const int rowBase = blockIdx.y * 64;
  const int colBase = blockIdx.x * 128;

  const char* Ab = (const char*)Abf;
  const char* Bb = (const char*)Wt;
  const unsigned laBuf[2] = { (unsigned)(uintptr_t)&Asp[0][0],
                              (unsigned)(uintptr_t)&Asp[1][0] };
  const unsigned lbBuf[2] = { (unsigned)(uintptr_t)&Bsp[0][0],
                              (unsigned)(uintptr_t)&Bsp[1][0] };

  v8f acc[2][4] = {};

  async_copy_tile(Ab, Bb, laBuf[0], lbBuf[0], t, rowBase, colBase, 0);

  for (int kk = 0; kk < KDIM / 32; ++kk) {
    const int cur = kk & 1;
    if (kk < KDIM / 32 - 1) {
      async_copy_tile(Ab, Bb, laBuf[cur ^ 1], lbBuf[cur ^ 1],
                      t, rowBase, colBase, (kk + 1) * 32);
      asm volatile("s_wait_asynccnt 0x6" ::: "memory");  // tile kk landed
    } else {
      asm volatile("s_wait_asynccnt 0x0" ::: "memory");
    }
    __syncthreads();   // all waves' copies for tile kk visible

    const unsigned int* Ac = Asp[cur];
    const unsigned int* Bc = Bsp[cur];

    v16bf a[2], b[4];
    #pragma unroll
    for (int mt = 0; mt < 2; ++mt) {
      const unsigned int* ar = Ac + (wm * 32 + mt * 16 + l15) * LDST;
      a[mt] = frag_from(ar + half * 4, ar + 8 + half * 4);
    }
    #pragma unroll
    for (int nt = 0; nt < 4; ++nt) {
      const unsigned int* br = Bc + (wn * 64 + nt * 16 + l15) * LDST;
      b[nt] = frag_from(br + half * 4, br + 8 + half * 4);
    }
    #pragma unroll
    for (int mt = 0; mt < 2; ++mt)
      #pragma unroll
      for (int nt = 0; nt < 4; ++nt)
        acc[mt][nt] = __builtin_amdgcn_wmma_f32_16x16x32_bf16(
            false, a[mt], false, b[nt], (short)0, acc[mt][nt], false, false);

    __syncthreads();   // tile kk fully consumed -> its buffer reusable
  }

  // Epilogue. D layout: lanes 0-15 -> N=l15, M rows 0..7; lanes 16-31 -> rows 8..15.
  #pragma unroll
  for (int nt = 0; nt < 4; ++nt) {
    const int n  = colBase + wn * 64 + nt * 16 + l15;
    const float bn = bias ? bias[n] : 0.f;
    #pragma unroll
    for (int mt = 0; mt < 2; ++mt) {
      const int mB = rowBase + wm * 32 + mt * 16 + half * 8;
      #pragma unroll
      for (int i = 0; i < 8; ++i) {
        const int m = mB + i;
        if (m < M) {
          float v = acc[mt][nt][i] + bn;
          if (R) v += R[(size_t)m * ldC + n];
          C[(size_t)m * ldC + n] = v;
        }
      }
    }
  }
}

// ----------------------------------------------------------------------------
// Softmax + multi-scale bilinear sampling. Block = 1 query, 8 wave32 = 8 heads.
// Lane doubles as sample-id j (phase 1) and channel d (phase 2).
// ----------------------------------------------------------------------------
__global__ __launch_bounds__(256)
void msda_sample(const float* __restrict__ V,   const float* __restrict__ OFF,
                 const float* __restrict__ ATT, const float* __restrict__ RP,
                 const int* __restrict__ shapes, const int* __restrict__ lsi,
                 unsigned short* __restrict__ TMPbf)
{
  __shared__ int   s_idx[HEADS][32][4];
  __shared__ float s_w  [HEADS][32][4];
  const int q    = blockIdx.x;
  const int h    = threadIdx.x >> 5;
  const int lane = threadIdx.x & 31;

  // softmax over the 32 (level,point) logits of this head
  float a  = ATT[(size_t)q * 256 + h * 32 + lane];
  float mx = a;
  for (int o = 16; o > 0; o >>= 1) mx = fmaxf(mx, __shfl_xor(mx, o, 32));
  float e = __expf(a - mx);
  float s = e;
  for (int o = 16; o > 0; o >>= 1) s += __shfl_xor(s, o, 32);
  const float wj = e / s;

  // phase 1: lane j = l*8+p computes 4 corner indices + (bilinear * attn) wts
  {
    const int l  = lane >> 3, p = lane & 7;
    const int Hl = shapes[l * 2 + 0];
    const int Wl = shapes[l * 2 + 1];
    const int base = lsi[l];
    const size_t ob = (size_t)q * 512 + (size_t)(((h * LEVELS + l) * POINTS + p) * 2);
    const float ox = OFF[ob + 0];
    const float oy = OFF[ob + 1];
    const float rx = RP[(size_t)q * 8 + l * 2 + 0];
    const float ry = RP[(size_t)q * 8 + l * 2 + 1];
    const float x = fmaf(rx, (float)Wl, ox) - 0.5f;   // loc*W - 0.5
    const float y = fmaf(ry, (float)Hl, oy) - 0.5f;
    const float x0f = floorf(x), y0f = floorf(y);
    const float fx = x - x0f,  fy = y - y0f;
    const int x0 = (int)x0f,   y0 = (int)y0f;
    #pragma unroll
    for (int c = 0; c < 4; ++c) {
      const int dy = c >> 1, dx = c & 1;
      const int xi = x0 + dx, yi = y0 + dy;
      const bool ok = (xi >= 0) && (xi < Wl) && (yi >= 0) && (yi < Hl);
      s_idx[h][lane][c] = ok ? (base + yi * Wl + xi) : -1;   // zero padding
      s_w  [h][lane][c] = (dy ? fy : 1.f - fy) * (dx ? fx : 1.f - fx) * wj;
    }
  }
  __syncthreads();

  // phase 2: lane = channel d; coalesced gathers of 32 consecutive floats
  float acc = 0.f;
  const int chan = h * HD + lane;
  for (int j = 0; j < 32; ++j) {
    #pragma unroll
    for (int c = 0; c < 4; ++c) {
      const int   idx = s_idx[h][j][c];   // uniform across wave -> broadcast
      const float w   = s_w[h][j][c];
      if (idx >= 0)
        acc = fmaf(w, V[(size_t)idx * EMBED + chan], acc);
    }
  }
  TMPbf[(size_t)q * EMBED + chan] = f2bf(acc);
}

// ----------------------------------------------------------------------------
extern "C" void kernel_launch(void* const* d_in, const int* in_sizes, int n_in,
                              void* d_out, int out_size, void* d_ws, size_t ws_size,
                              hipStream_t stream)
{
  const float* query   = (const float*)d_in[0];
  const float* value   = (const float*)d_in[1];
  const float* rp      = (const float*)d_in[2];
  const int*   shapes  = (const int*)  d_in[3];
  const int*   lsi     = (const int*)  d_in[4];
  const float* W_value = (const float*)d_in[5];
  const float* b_value = (const float*)d_in[6];
  const float* W_off   = (const float*)d_in[7];
  const float* b_off   = (const float*)d_in[8];
  const float* W_attn  = (const float*)d_in[9];
  const float* b_attn  = (const float*)d_in[10];
  const float* W_out   = (const float*)d_in[11];
  const float* b_out   = (const float*)d_in[12];

  const int nq    = in_sizes[0] / EMBED;            // 22500
  const int nv    = in_sizes[1] / EMBED;            // 13294
  const int nqPad = ((nq + 63) / 64) * 64;          // 22528
  const int nvPad = ((nv + 63) / 64) * 64;          // 13312

  char* w = (char*)d_ws;
  auto alloc = [&](size_t bytes) -> char* {
    char* p = w; w += (bytes + 255) & ~(size_t)255; return p;
  };
  float*          Vp    = (float*)         alloc((size_t)nv * EMBED * 4);
  float*          OFF   = (float*)         alloc((size_t)nq * 512   * 4);
  float*          ATT   = (float*)         alloc((size_t)nq * EMBED * 4);
  unsigned int*   Qbf   = (unsigned int*)  alloc((size_t)nqPad * EMBED * 2);
  unsigned int*   Vbf   = (unsigned int*)  alloc((size_t)nvPad * EMBED * 2);
  unsigned int*   Tbf   = (unsigned int*)  alloc((size_t)nqPad * EMBED * 2);
  unsigned int*   Wv_t  = (unsigned int*)  alloc((size_t)256 * EMBED * 2);
  unsigned int*   Woff_t= (unsigned int*)  alloc((size_t)512 * EMBED * 2);
  unsigned int*   Watt_t= (unsigned int*)  alloc((size_t)256 * EMBED * 2);
  unsigned int*   Wout_t= (unsigned int*)  alloc((size_t)256 * EMBED * 2);

  // --- precision / layout conversion -------------------------------------
  {
    long long pq = (long long)nqPad * 128, pv = (long long)nvPad * 128;
    cvt_act_bf16<<<dim3((unsigned)((pq + 255) / 256)), 256, 0, stream>>>(query, Qbf, nq, pq);
    cvt_act_bf16<<<dim3((unsigned)((pv + 255) / 256)), 256, 0, stream>>>(value, Vbf, nv, pv);
    cvt_w_bf16<<<dim3((256 * 128) / 256), 256, 0, stream>>>(W_value, Wv_t, 256);
    cvt_w_bf16<<<dim3((512 * 128) / 256), 256, 0, stream>>>(W_off,  Woff_t, 512);
    cvt_w_bf16<<<dim3((256 * 128) / 256), 256, 0, stream>>>(W_attn, Watt_t, 256);
    cvt_w_bf16<<<dim3((256 * 128) / 256), 256, 0, stream>>>(W_out,  Wout_t, 256);
    int padPairs = (nqPad - nq) * 128;
    if (padPairs > 0)
      zero_pairs<<<dim3((padPairs + 255) / 256), 256, 0, stream>>>(
          Tbf + (size_t)nq * 128, padPairs);
  }

  const dim3 blk(128);
  // 1. value projection: v = value @ W_value + b_value          (N=256)
  gemm_bf16wmma_async<<<dim3(2, nvPad / 64), blk, 0, stream>>>(
      (const unsigned short*)Vbf, (const unsigned short*)Wv_t,
      b_value, nullptr, Vp, nv, EMBED);
  // 2. sampling offsets: OFF = query @ W_off + b_off            (N=512)
  gemm_bf16wmma_async<<<dim3(4, nqPad / 64), blk, 0, stream>>>(
      (const unsigned short*)Qbf, (const unsigned short*)Woff_t,
      b_off, nullptr, OFF, nq, 512);
  // 3. attention logits: ATT = query @ W_attn + b_attn          (N=256)
  gemm_bf16wmma_async<<<dim3(2, nqPad / 64), blk, 0, stream>>>(
      (const unsigned short*)Qbf, (const unsigned short*)Watt_t,
      b_attn, nullptr, ATT, nq, EMBED);
  // 4. softmax + bilinear gather + weighted sum -> bf16
  msda_sample<<<dim3(nq), dim3(256), 0, stream>>>(
      Vp, OFF, ATT, rp, shapes, lsi, (unsigned short*)Tbf);
  // 5. output projection + residual: out = TMP @ W_out + b_out + query
  gemm_bf16wmma_async<<<dim3(2, nqPad / 64), blk, 0, stream>>>(
      (const unsigned short*)Tbf, (const unsigned short*)Wout_t,
      b_out, query, (float*)d_out, nq, EMBED);
}
